// MaskedAdaAttN_72765335929382
// MI455X (gfx1250) — matrix operand order repaired and checked by reading, not compiled, not run on hardware
//
#include <hip/hip_runtime.h>
#include <hip/hip_bf16.h>

typedef __attribute__((ext_vector_type(16))) __bf16 v16bf;
typedef __attribute__((ext_vector_type(8)))  float  v8f;

#define BS 2
#define C  1024
#define T  2048
#define H  16
#define D  64

// ---- WMMA fragment loaders (CDNA5 16-bit layouts, LDS rows hold 32 bf16 of K) ----
// A (16x32, MxK): lanes 0-15 -> K 0..7 / 16..23 ; lanes 16-31 -> K 8..15 / 24..31
__device__ inline v16bf frag_a(const __bf16* row, int lane) {
  const uint4* p = (const uint4*)row;
  int h = (lane >> 4) & 1;
  v16bf f;
  ((uint4*)&f)[0] = p[h];
  ((uint4*)&f)[1] = p[2 + h];
  return f;
}
// B (32x16, KxN) stored as [n][k]: lanes 0-15 -> K 0..7 / 16..23 ; lanes 16-31 -> K 8..15 / 24..31
__device__ inline v16bf frag_b(const __bf16* row, int lane) {
  const uint4* p = (const uint4*)row;
  int h = (lane >> 4) & 1;
  v16bf f;
  ((uint4*)&f)[0] = p[h];
  ((uint4*)&f)[1] = p[2 + h];
  return f;
}

// =====================  projection GEMM:  O = (W @ X + b) * s  (bf16 out) =====================
__global__ __launch_bounds__(256) void proj_gemm(
    const float* __restrict__ W, const float* __restrict__ Xall,
    const float* __restrict__ bias, __bf16* __restrict__ Oall, float outscale)
{
  __shared__ __align__(16) __bf16 Wt[64][32];    // [o_local][k]
  __shared__ __align__(16) __bf16 Xt[128][32];   // [t_local][k] (transposed)

  const int tid  = threadIdx.x;
  const int lane = tid & 31;
  const int w    = tid >> 5;
  const int waveo = w >> 2;            // 0..1
  const int wavet = w & 3;             // 0..3
  const int o0 = blockIdx.y * 64;
  const int t0 = blockIdx.x * 128;
  const float* X = Xall + (size_t)blockIdx.z * C * T;
  __bf16*      O = Oall + (size_t)blockIdx.z * C * T;

  v8f acc[2][2] = {};

  const int wrow = tid >> 2;           // 0..63
  const int wcol = (tid & 3) * 8;      // 0,8,16,24
  const int xk   = tid >> 3;           // 0..31
  const int xt0  = (tid & 7) * 16;     // 0..112

  for (int k0 = 0; k0 < C; k0 += 32) {
    // prefetch next K-chunk a full tile ahead (uniform branch; speculative, dropped if OOB)
    if (k0 + 32 < C) {
      __builtin_prefetch(W + (size_t)(o0 + wrow) * C + k0 + 32 + wcol, 0, 1);
      __builtin_prefetch(X + (size_t)(k0 + 32 + xk) * T + t0 + xt0, 0, 1);
    }
    __syncthreads();
    { // stage W 64x32 -> bf16
      const float* src = W + (size_t)(o0 + wrow) * C + k0 + wcol;
      float4 f0 = ((const float4*)src)[0];
      float4 f1 = ((const float4*)src)[1];
      __align__(16) __bf16 tmp[8];
      tmp[0]=(__bf16)f0.x; tmp[1]=(__bf16)f0.y; tmp[2]=(__bf16)f0.z; tmp[3]=(__bf16)f0.w;
      tmp[4]=(__bf16)f1.x; tmp[5]=(__bf16)f1.y; tmp[6]=(__bf16)f1.z; tmp[7]=(__bf16)f1.w;
      *(uint4*)&Wt[wrow][wcol] = *(const uint4*)tmp;
    }
    { // stage X 32x128 -> transposed [t][k] bf16
      const float* src = X + (size_t)(k0 + xk) * T + t0 + xt0;
      for (int j = 0; j < 4; ++j) {
        float4 f = ((const float4*)src)[j];
        Xt[xt0 + 4*j + 0][xk] = (__bf16)f.x;
        Xt[xt0 + 4*j + 1][xk] = (__bf16)f.y;
        Xt[xt0 + 4*j + 2][xk] = (__bf16)f.z;
        Xt[xt0 + 4*j + 3][xk] = (__bf16)f.w;
      }
    }
    __syncthreads();

    v16bf a0 = frag_a(&Wt[waveo*32 +  0 + (lane&15)][0], lane);
    v16bf a1 = frag_a(&Wt[waveo*32 + 16 + (lane&15)][0], lane);
    v16bf b0 = frag_b(&Xt[wavet*32 +  0 + (lane&15)][0], lane);
    v16bf b1 = frag_b(&Xt[wavet*32 + 16 + (lane&15)][0], lane);
    acc[0][0] = __builtin_amdgcn_wmma_f32_16x16x32_bf16(false, a0, false, b0, (short)0, acc[0][0], false, false);
    acc[0][1] = __builtin_amdgcn_wmma_f32_16x16x32_bf16(false, a0, false, b1, (short)0, acc[0][1], false, false);
    acc[1][0] = __builtin_amdgcn_wmma_f32_16x16x32_bf16(false, a1, false, b0, (short)0, acc[1][0], false, false);
    acc[1][1] = __builtin_amdgcn_wmma_f32_16x16x32_bf16(false, a1, false, b1, (short)0, acc[1][1], false, false);
  }

  const int mb = (lane & 16) ? 8 : 0;      // C/D layout: lanes 16-31 hold rows M+8
  for (int mt = 0; mt < 2; ++mt)
    for (int nt = 0; nt < 2; ++nt)
      for (int r = 0; r < 8; ++r) {
        int og = o0 + waveo*32 + mt*16 + mb + r;
        int tg = t0 + wavet*32 + nt*16 + (lane & 15);
        float v = acc[mt][nt][r] + bias[og];
        O[(size_t)og * T + tg] = (__bf16)(v * outscale);
      }
}

// =====================  flash attention producing mean = P@V, std = sqrt(P@V^2 - mean^2) ==========
__global__ __launch_bounds__(256) void attn_kernel(
    const __bf16* __restrict__ Qall, const __bf16* __restrict__ Kall,
    const __bf16* __restrict__ Vall, const unsigned char* __restrict__ kvmask,
    float* __restrict__ meanO, float* __restrict__ stdO)
{
  __shared__ __align__(16) __bf16 Qs[128][64];     // [q_local][d]
  __shared__ __align__(16) __bf16 Ks[32][64];      // [key_local][d]
  __shared__ __align__(16) __bf16 Vs[64][32];      // [d][key_local]
  __shared__ __align__(16) __bf16 V2s[64][32];     // squared
  __shared__ __align__(16) __bf16 Ps[8][16][32];   // per-wave P tile (C->A relayout)

  const int tid = threadIdx.x, lane = tid & 31, w = tid >> 5;
  const int b = blockIdx.z, h = blockIdx.y;
  const int q0 = blockIdx.x * 128;
  const __bf16* Q = Qall + (size_t)b * C * T + (size_t)h * D * T;
  const __bf16* K = Kall + (size_t)b * C * T + (size_t)h * D * T;
  const __bf16* V = Vall + (size_t)b * C * T + (size_t)h * D * T;
  const unsigned char* msk = kvmask + (size_t)b * T;

  { // stage Q tile 128x64 (transpose from [d][t] global)
    int kd = tid >> 2;            // 0..63
    int m0 = (tid & 3) * 32;      // 0,32,64,96
    const __bf16* src = Q + (size_t)kd * T + q0 + m0;
    for (int j = 0; j < 32; ++j) Qs[m0 + j][kd] = src[j];
  }
  __syncthreads();

  v16bf qa0 = frag_a(&Qs[w*16 + (lane&15)][0],  lane);   // K-dim 0..31
  v16bf qa1 = frag_a(&Qs[w*16 + (lane&15)][32], lane);   // K-dim 32..63

  float m_run[8], l_run[8];
  for (int r = 0; r < 8; ++r) { m_run[r] = -1e30f; l_run[r] = 0.f; }
  v8f accM[4] = {}, accE[4] = {};

  const int skd = tid >> 2;           // 0..63
  const int sk0 = (tid & 3) * 8;      // 0,8,16,24
  const int mb  = (lane & 16) ? 8 : 0;

  // LDS byte offset of this thread's V-chunk slot: on CDNA5 the shared aperture
  // occupies ADDR[63:32], so the low 32 bits of a flat LDS address are the LDS offset.
  const unsigned vs_lds_off = (unsigned)(unsigned long long)(const void*)&Vs[skd][sk0];

  for (int kc = 0; kc < T; kc += 32) {
    if (kc + 32 < T) { // prefetch next chunk (uniform branch)
      __builtin_prefetch(K + (size_t)skd * T + kc + 32 + sk0, 0, 1);
      __builtin_prefetch(V + (size_t)skd * T + kc + 32 + sk0, 0, 1);
    }
    __syncthreads();

    { // async-copy V chunk L2->LDS (natural [d][key] layout, no VGPR round-trip)
      const __bf16* gsrc = V + (size_t)skd * T + kc + sk0;
      asm volatile("global_load_async_to_lds_b128 %0, %1, off"
                   :: "v"(vs_lds_off), "v"(gsrc) : "memory");
    }
    { // meanwhile stage K chunk transposed -> Ks[key][d] through VGPRs
      const __bf16* src = K + (size_t)skd * T + kc + sk0;
      uint4 raw = *(const uint4*)src;
      const __bf16* e = (const __bf16*)&raw;
      for (int j = 0; j < 8; ++j) Ks[sk0 + j][skd] = e[j];
    }
    // wait for this wave's async V copy, then derive V^2 from the landed tile
    asm volatile("s_wait_asynccnt 0x0" ::: "memory");
    {
      uint4 raw = *(const uint4*)&Vs[skd][sk0];
      const __bf16* e = (const __bf16*)&raw;
      __align__(16) __bf16 sq[8];
      for (int j = 0; j < 8; ++j) { float f = (float)e[j]; sq[j] = (__bf16)(f * f); }
      *(uint4*)&V2s[skd][sk0] = *(const uint4*)sq;
    }
    __syncthreads();

    // S(16x32) = Q(16x64) x K^T(64x32)   (scale pre-folded into Q)
    v8f s0 = {}, s1 = {};
    {
      v16bf kb00 = frag_b(&Ks[ 0 + (lane&15)][0],  lane);
      v16bf kb01 = frag_b(&Ks[ 0 + (lane&15)][32], lane);
      v16bf kb10 = frag_b(&Ks[16 + (lane&15)][0],  lane);
      v16bf kb11 = frag_b(&Ks[16 + (lane&15)][32], lane);
      s0 = __builtin_amdgcn_wmma_f32_16x16x32_bf16(false, qa0, false, kb00, (short)0, s0, false, false);
      s0 = __builtin_amdgcn_wmma_f32_16x16x32_bf16(false, qa1, false, kb01, (short)0, s0, false, false);
      s1 = __builtin_amdgcn_wmma_f32_16x16x32_bf16(false, qa0, false, kb10, (short)0, s1, false, false);
      s1 = __builtin_amdgcn_wmma_f32_16x16x32_bf16(false, qa1, false, kb11, (short)0, s1, false, false);
    }

    // kv mask (this lane's two key columns)
    bool ok0 = msk[kc + (lane & 15)] != 0;
    bool ok1 = msk[kc + 16 + (lane & 15)] != 0;

    // online softmax; row r lives in VGPR r of this lane's half -> stats reduce across 16 lanes via xor shuffles
    for (int r = 0; r < 8; ++r) {
      float a = ok0 ? s0[r] : -1e30f;
      float c = ok1 ? s1[r] : -1e30f;
      float v = fmaxf(a, c);
      for (int off = 1; off < 16; off <<= 1)
        v = fmaxf(v, __shfl_xor(v, off, 32));
      float newm = fmaxf(m_run[r], v);
      float f = __expf(m_run[r] - newm);
      m_run[r] = newm;
      float p0 = __expf(a - newm);
      float p1 = __expf(c - newm);
      float rs = p0 + p1;
      for (int off = 1; off < 16; off <<= 1)
        rs += __shfl_xor(rs, off, 32);
      l_run[r] = l_run[r] * f + rs;
      for (int j = 0; j < 4; ++j) { accM[j][r] *= f; accE[j][r] *= f; }
      // re-layout C->A via per-wave LDS tile
      Ps[w][mb + r][lane & 15]        = (__bf16)p0;
      Ps[w][mb + r][16 + (lane & 15)] = (__bf16)p1;
    }

    v16bf pf = frag_a(&Ps[w][lane & 15][0], lane);
    for (int j = 0; j < 4; ++j) {
      v16bf vf  = frag_b(&Vs[j*16 + (lane&15)][0],  lane);
      v16bf v2f = frag_b(&V2s[j*16 + (lane&15)][0], lane);
      accM[j] = __builtin_amdgcn_wmma_f32_16x16x32_bf16(false, pf, false, vf,  (short)0, accM[j], false, false);
      accE[j] = __builtin_amdgcn_wmma_f32_16x16x32_bf16(false, pf, false, v2f, (short)0, accE[j], false, false);
    }
  }

  for (int r = 0; r < 8; ++r) {
    float inv = 1.0f / fmaxf(l_run[r], 1e-20f);
    int tg = q0 + w*16 + mb + r;
    for (int j = 0; j < 4; ++j) {
      float mu = accM[j][r] * inv;
      float e2 = accE[j][r] * inv;
      float sd = sqrtf(fmaxf(e2 - mu * mu, 0.f));
      int ch = h * D + j * 16 + (lane & 15);
      size_t idx = ((size_t)b * C + ch) * T + tg;
      meanO[idx] = mu;
      stdO[idx]  = sd;
    }
  }
}

// =====================  masked instance norm + combine  =====================
__global__ __launch_bounds__(256) void finalize_kernel(
    const float* __restrict__ Xall, const unsigned char* __restrict__ maskall,
    const float* __restrict__ meanA, const float* __restrict__ stdA,
    float* __restrict__ out)
{
  __shared__ float rs[256], rq[256], rc[256];
  const int tid = threadIdx.x;
  const int row = blockIdx.x;                // b*C + ch
  const int b   = row / C;
  const float* x = Xall + (size_t)row * T;
  const unsigned char* m = maskall + (size_t)b * T;

  float s = 0.f, q = 0.f, c = 0.f;
  for (int i = tid; i < T; i += 256) {
    float mv = m[i] ? 1.f : 0.f;
    float xv = x[i] * mv;
    s += xv; q += xv * xv; c += mv;
  }
  rs[tid] = s; rq[tid] = q; rc[tid] = c;
  __syncthreads();
  for (int off = 128; off > 0; off >>= 1) {
    if (tid < off) { rs[tid] += rs[tid+off]; rq[tid] += rq[tid+off]; rc[tid] += rc[tid+off]; }
    __syncthreads();
  }
  float n   = fmaxf(rc[0], 1.f);
  float mu  = rs[0] / n;
  float var = rq[0] / n - mu * mu;
  float inv = rsqrtf(fmaxf(var, 0.f) + 1e-5f);

  for (int i = tid; i < T; i += 256) {
    float mv = m[i] ? 1.f : 0.f;
    float xn = (x[i] * mv - mu) * inv * mv;
    size_t idx = (size_t)row * T + i;
    out[idx] = (xn * stdA[idx] + meanA[idx]) * mv;
  }
}

extern "C" void kernel_launch(void* const* d_in, const int* in_sizes, int n_in,
                              void* d_out, int out_size, void* d_ws, size_t ws_size,
                              hipStream_t stream)
{
  const float*         x    = (const float*)d_in[0];
  const unsigned char* mask = (const unsigned char*)d_in[1];
  const float*         kv   = (const float*)d_in[2];
  const unsigned char* kvm  = (const unsigned char*)d_in[3];
  const float* Wq = (const float*)d_in[4];
  const float* bq = (const float*)d_in[5];
  const float* Wk = (const float*)d_in[6];
  const float* bk = (const float*)d_in[7];
  const float* Wv = (const float*)d_in[8];
  const float* bv = (const float*)d_in[9];

  const size_t nElem = (size_t)BS * C * T;
  __bf16* qb = (__bf16*)d_ws;
  __bf16* kb = qb + nElem;
  __bf16* vb = kb + nElem;
  float* meanA = (float*)(vb + nElem);
  float* stdA  = meanA + nElem;

  dim3 blk(256);
  dim3 gG(T / 128, C / 64, BS);
  proj_gemm<<<gG, blk, 0, stream>>>(Wq, x,  bq, qb, 0.125f);  // 1/sqrt(64) folded into Q
  proj_gemm<<<gG, blk, 0, stream>>>(Wk, kv, bk, kb, 1.0f);
  proj_gemm<<<gG, blk, 0, stream>>>(Wv, kv, bv, vb, 1.0f);

  dim3 gA(T / 128, H, BS);
  attn_kernel<<<gA, blk, 0, stream>>>(qb, kb, vb, kvm, meanA, stdA);

  finalize_kernel<<<BS * C, blk, 0, stream>>>(x, mask, meanA, stdA, (float*)d_out);
}
